// Attention_39032662786718
// MI455X (gfx1250) — matrix operand order repaired and checked by reading, not compiled, not run on hardware
//
#include <hip/hip_runtime.h>

#define N_CTX   8192
#define DMODEL  1024
#define BM      16      // Q rows per workgroup
#define BN      128     // KV rows per iteration (8 waves x 16)
#define WAVES   8
#define DSLICE  128     // output columns owned by each wave
#define QS_STRIDE 1032  // 1024 + 8 pad (bf16 elems) -> 2064B row, 16B aligned
#define PS_STRIDE 136   // 128 + 8 pad  (bf16 elems) -> 272B  row, 16B aligned

typedef __bf16 v2bf  __attribute__((ext_vector_type(2)));
typedef __bf16 v8bf  __attribute__((ext_vector_type(8)));
typedef __bf16 v16bf __attribute__((ext_vector_type(16)));
typedef float  v8f   __attribute__((ext_vector_type(8)));

static __device__ __forceinline__ v2bf cvt2bf(float a, float b) {
#if __has_builtin(__builtin_amdgcn_cvt_pk_bf16_f32)
    return __builtin_amdgcn_cvt_pk_bf16_f32(a, b);
#else
    unsigned ua = __builtin_bit_cast(unsigned, a);
    unsigned ub = __builtin_bit_cast(unsigned, b);
    ua = ua + 0x7FFFu + ((ua >> 16) & 1u);
    ub = ub + 0x7FFFu + ((ub >> 16) & 1u);
    unsigned r = (ua >> 16) | (ub & 0xFFFF0000u);
    return __builtin_bit_cast(v2bf, r);
#endif
}

static __device__ __forceinline__ __bf16 f2bf(float f) {
    unsigned u = __builtin_bit_cast(unsigned, f);
    unsigned r = u + 0x7FFFu + ((u >> 16) & 1u);
    unsigned short h = (unsigned short)(r >> 16);
    return __builtin_bit_cast(__bf16, h);
}

static __device__ __forceinline__ float fast_exp2(float x) {
#if __has_builtin(__builtin_amdgcn_exp2f)
    return __builtin_amdgcn_exp2f(x);
#else
    return exp2f(x);
#endif
}

// ============================================================================
// Prologue 1: K fp32 -> bf16, same row-major layout
// ============================================================================
__global__ void __launch_bounds__(256)
convert_k_bf16(const float* __restrict__ K, __bf16* __restrict__ Kb)
{
    const size_t i = ((size_t)blockIdx.x * 256 + threadIdx.x) * 8;
    const float4 a = *(const float4*)(K + i);
    const float4 b = *(const float4*)(K + i + 4);
    uint4 st;
    st.x = __builtin_bit_cast(unsigned, cvt2bf(a.x, a.y));
    st.y = __builtin_bit_cast(unsigned, cvt2bf(a.z, a.w));
    st.z = __builtin_bit_cast(unsigned, cvt2bf(b.x, b.y));
    st.w = __builtin_bit_cast(unsigned, cvt2bf(b.z, b.w));
    *(uint4*)(Kb + i) = st;
}

// ============================================================================
// Prologue 2: V fp32 [n][d] -> Vt bf16 [d][n]  (LDS-tiled 64x64 transpose)
// ============================================================================
__global__ void __launch_bounds__(256)
transpose_v_bf16(const float* __restrict__ V, __bf16* __restrict__ Vt)
{
    __shared__ alignas(16) __bf16 Lt[64 * 72];    // [d][n], padded stride
    const int t  = threadIdx.x;
    const int n0 = (blockIdx.x & 127) * 64;       // 128 tiles along n
    const int d0 = (blockIdx.x >> 7) * 64;        // 16 tiles along d
    #pragma unroll
    for (int i = 0; i < 4; ++i) {
        const int idx = t + i * 256;              // 0..1023
        const int r   = idx >> 4;                 // n offset 0..63
        const int c4  = idx & 15;                 // float4 chunk along d
        const float4 f = *(const float4*)(V + (size_t)(n0 + r) * DMODEL + d0 + c4 * 4);
        Lt[(c4 * 4 + 0) * 72 + r] = f2bf(f.x);
        Lt[(c4 * 4 + 1) * 72 + r] = f2bf(f.y);
        Lt[(c4 * 4 + 2) * 72 + r] = f2bf(f.z);
        Lt[(c4 * 4 + 3) * 72 + r] = f2bf(f.w);
    }
    __syncthreads();
    #pragma unroll
    for (int i = 0; i < 2; ++i) {
        const int idx = t + i * 256;              // 0..511
        const int d   = idx >> 3;                 // 0..63
        const int q   = idx & 7;                  // 8-elem chunk along n
        *(v8bf*)(Vt + (size_t)(d0 + d) * N_CTX + n0 + q * 8) =
            *(const v8bf*)(Lt + d * 72 + q * 8);
    }
}

// ============================================================================
// Main flash-attention kernel, bf16 K (row-major) + bf16 V^T operands
// ============================================================================
__global__ void __launch_bounds__(256)
fattn_fwd_bf16(const float* __restrict__ Q, const __bf16* __restrict__ Kb,
               const __bf16* __restrict__ Vt, float* __restrict__ O)
{
    __shared__ alignas(16) __bf16 Qs[BM * QS_STRIDE];
    __shared__ alignas(16) __bf16 Pb[BM * PS_STRIDE];
    __shared__ float mw[WAVES * 16];
    __shared__ float sw[WAVES * 16];
    __shared__ alignas(16) float bcst[WAVES * 32];

    const int tid   = threadIdx.x;
    const int lane  = tid & 31;
    const int wave  = tid >> 5;
    const int ln    = lane & 15;
    const int hi    = lane >> 4;
    const int qbase = blockIdx.x * BM;

    // stage Q into LDS as bf16, prescaled by log2(e)/sqrt(D): log2-domain scores
    {
        const float s = 0.04508422f;              // log2(e) / 32
        for (int idx = tid; idx < BM * (DMODEL / 4); idx += 256) {
            const int row = idx >> 8;
            const int c4  = idx & 255;
            const float4 q4 = *(const float4*)(Q + (size_t)(qbase + row) * DMODEL + (size_t)c4 * 4);
            uint2 st;
            st.x = __builtin_bit_cast(unsigned, cvt2bf(q4.x * s, q4.y * s));
            st.y = __builtin_bit_cast(unsigned, cvt2bf(q4.z * s, q4.w * s));
            *(uint2*)(Qs + row * QS_STRIDE + c4 * 4) = st;
        }
    }

    float m_run = -3.0e38f;
    float l_run = 0.0f;
    v8f o[8];
    #pragma unroll
    for (int t = 0; t < 8; ++t) o[t] = (v8f){0.f,0.f,0.f,0.f,0.f,0.f,0.f,0.f};

    const __bf16* vt0 = Vt + (size_t)(wave * DSLICE + ln) * N_CTX;   // per-lane column base

    for (int it = 0; it < N_CTX / BN; ++it) {
        __syncthreads();                          // A: Qs ready / Pb,stats consumed
        const int nb = it * BN;

        // ============ S = Q @ K_slice^T (16x16, K-dim 1024) ============
        v8f c = (v8f){0.f,0.f,0.f,0.f,0.f,0.f,0.f,0.f};
        const __bf16* krb = Kb + (size_t)(nb + wave * 16 + ln) * DMODEL;
        if (it + 1 < N_CTX / BN) {
            __builtin_prefetch(Kb + (size_t)(nb + BN + wave * 16 + ln) * DMODEL, 0, 1);
            __builtin_prefetch(vt0 + nb + BN, 0, 1);
        }

        for (int ks = 0; ks < DMODEL / 32; ++ks) {
            const int k0 = ks * 32;
            union { v8bf h[2]; v16bf v; } au, bu;
            // A fragment (Q) from LDS
            au.h[0] = *(const v8bf*)(Qs + ln * QS_STRIDE + k0 + 16 * hi);
            au.h[1] = *(const v8bf*)(Qs + ln * QS_STRIDE + k0 + 16 * hi + 8);
            // B fragment (K^T): lane = KV row, 16 contiguous bf16
            const __bf16* kp = krb + k0 + 16 * hi;
            bu.h[0] = *(const v8bf*)(kp);
            bu.h[1] = *(const v8bf*)(kp + 8);
            c = __builtin_amdgcn_wmma_f32_16x16x32_bf16(false, au.v, false, bu.v,
                                                        (short)0, c, false, false);
        }

        // ============ online softmax (log2 domain) ============
        #pragma unroll
        for (int r = 0; r < 8; ++r) {
            float v = c[r];
            v = fmaxf(v, __shfl_xor(v, 1, 32));
            v = fmaxf(v, __shfl_xor(v, 2, 32));
            v = fmaxf(v, __shfl_xor(v, 4, 32));
            v = fmaxf(v, __shfl_xor(v, 8, 32));
            mw[wave * 16 + r + 8 * hi] = v;
        }
        __syncthreads();                          // B

        float mnew = m_run;
        #pragma unroll
        for (int w2 = 0; w2 < WAVES; ++w2) mnew = fmaxf(mnew, mw[w2 * 16 + ln]);
        const float alpha = fast_exp2(m_run - mnew);
        m_run = mnew;
        bcst[wave * 32 + ln]      = mnew;
        bcst[wave * 32 + 16 + ln] = alpha;
        float mnb[8], alb[8];
        {
            const float4 x0 = *(const float4*)(bcst + wave * 32 + 8 * hi);
            const float4 x1 = *(const float4*)(bcst + wave * 32 + 8 * hi + 4);
            mnb[0]=x0.x; mnb[1]=x0.y; mnb[2]=x0.z; mnb[3]=x0.w;
            mnb[4]=x1.x; mnb[5]=x1.y; mnb[6]=x1.z; mnb[7]=x1.w;
            const float4 y0 = *(const float4*)(bcst + wave * 32 + 16 + 8 * hi);
            const float4 y1 = *(const float4*)(bcst + wave * 32 + 16 + 8 * hi + 4);
            alb[0]=y0.x; alb[1]=y0.y; alb[2]=y0.z; alb[3]=y0.w;
            alb[4]=y1.x; alb[5]=y1.y; alb[6]=y1.z; alb[7]=y1.w;
        }

        #pragma unroll
        for (int r = 0; r < 8; ++r) {
            const float p = fast_exp2(c[r] - mnb[r]);
            Pb[(r + 8 * hi) * PS_STRIDE + wave * 16 + ln] = f2bf(p);
            float v = p;
            v += __shfl_xor(v, 1, 32);
            v += __shfl_xor(v, 2, 32);
            v += __shfl_xor(v, 4, 32);
            v += __shfl_xor(v, 8, 32);
            sw[wave * 16 + r + 8 * hi] = v;
        }
        #pragma unroll
        for (int t = 0; t < 8; ++t) {
            #pragma unroll
            for (int r = 0; r < 8; ++r) o[t][r] *= alb[r];
        }
        __syncthreads();                          // D: Pb + sums visible

        float lsum = 0.f;
        #pragma unroll
        for (int w2 = 0; w2 < WAVES; ++w2) lsum += sw[w2 * 16 + ln];
        l_run = alpha * l_run + lsum;

        // ============ O += P @ V (per-wave 128-column slice) ============
        #pragma unroll
        for (int kf = 0; kf < 4; ++kf) {
            union { v8bf h[2]; v16bf v; } au;
            au.h[0] = *(const v8bf*)(Pb + ln * PS_STRIDE + kf * 32 + 16 * hi);
            au.h[1] = *(const v8bf*)(Pb + ln * PS_STRIDE + kf * 32 + 16 * hi + 8);
            #pragma unroll
            for (int t = 0; t < 8; ++t) {
                // B fragment (V): lane = output column, 16 contiguous bf16 along kv
                const __bf16* vp = vt0 + (size_t)(t * 16) * N_CTX + nb + kf * 32 + 16 * hi;
                union { v8bf h[2]; v16bf v; } bu;
                bu.h[0] = *(const v8bf*)(vp);
                bu.h[1] = *(const v8bf*)(vp + 8);
                o[t] = __builtin_amdgcn_wmma_f32_16x16x32_bf16(false, au.v, false, bu.v,
                                                               (short)0, o[t], false, false);
            }
        }
    }

    const float inv = 1.0f / l_run;
    bcst[wave * 32 + ln] = inv;
    float invb[8];
    {
        const float4 z0 = *(const float4*)(bcst + wave * 32 + 8 * hi);
        const float4 z1 = *(const float4*)(bcst + wave * 32 + 8 * hi + 4);
        invb[0]=z0.x; invb[1]=z0.y; invb[2]=z0.z; invb[3]=z0.w;
        invb[4]=z1.x; invb[5]=z1.y; invb[6]=z1.z; invb[7]=z1.w;
    }
    #pragma unroll
    for (int t = 0; t < 8; ++t) {
        #pragma unroll
        for (int r = 0; r < 8; ++r) {
            O[(size_t)(qbase + r + 8 * hi) * DMODEL + wave * DSLICE + t * 16 + ln]
                = o[t][r] * invb[r];
        }
    }
}

// ============================================================================
// Fallback (no workspace): fp32 operands converted on the fly (round-2 kernel)
// ============================================================================
__global__ void __launch_bounds__(256)
fattn_fwd_f32(const float* __restrict__ Q, const float* __restrict__ K,
              const float* __restrict__ V, float* __restrict__ O)
{
    __shared__ alignas(16) __bf16 Qs[BM * QS_STRIDE];
    __shared__ alignas(16) __bf16 Pb[BM * PS_STRIDE];
    __shared__ float mw[WAVES * 16];
    __shared__ float sw[WAVES * 16];
    __shared__ alignas(16) float bcst[WAVES * 32];

    const int tid   = threadIdx.x;
    const int lane  = tid & 31;
    const int wave  = tid >> 5;
    const int ln    = lane & 15;
    const int hi    = lane >> 4;
    const int qbase = blockIdx.x * BM;

    {
        const float s = 0.04508422f;
        for (int idx = tid; idx < BM * (DMODEL / 4); idx += 256) {
            const int row = idx >> 8;
            const int c4  = idx & 255;
            const float4 q4 = *(const float4*)(Q + (size_t)(qbase + row) * DMODEL + (size_t)c4 * 4);
            uint2 st;
            st.x = __builtin_bit_cast(unsigned, cvt2bf(q4.x * s, q4.y * s));
            st.y = __builtin_bit_cast(unsigned, cvt2bf(q4.z * s, q4.w * s));
            *(uint2*)(Qs + row * QS_STRIDE + c4 * 4) = st;
        }
    }

    float m_run = -3.0e38f;
    float l_run = 0.0f;
    v8f o[8];
    #pragma unroll
    for (int t = 0; t < 8; ++t) o[t] = (v8f){0.f,0.f,0.f,0.f,0.f,0.f,0.f,0.f};

    for (int it = 0; it < N_CTX / BN; ++it) {
        __syncthreads();
        const int nb = it * BN;

        v8f c = (v8f){0.f,0.f,0.f,0.f,0.f,0.f,0.f,0.f};
        const float* kr = K + (size_t)(nb + wave * 16 + ln) * DMODEL;

        for (int ks = 0; ks < DMODEL / 32; ++ks) {
            const int k0 = ks * 32;
            union { v8bf h[2]; v16bf v; } au;
            au.h[0] = *(const v8bf*)(Qs + ln * QS_STRIDE + k0 + 16 * hi);
            au.h[1] = *(const v8bf*)(Qs + ln * QS_STRIDE + k0 + 16 * hi + 8);
            const float4* kp = (const float4*)(kr + k0 + 16 * hi);
            const float4 f0 = kp[0], f1 = kp[1], f2v = kp[2], f3 = kp[3];
            union { v2bf p[8]; v16bf v; } bu;
            bu.p[0] = cvt2bf(f0.x,  f0.y);  bu.p[1] = cvt2bf(f0.z,  f0.w);
            bu.p[2] = cvt2bf(f1.x,  f1.y);  bu.p[3] = cvt2bf(f1.z,  f1.w);
            bu.p[4] = cvt2bf(f2v.x, f2v.y); bu.p[5] = cvt2bf(f2v.z, f2v.w);
            bu.p[6] = cvt2bf(f3.x,  f3.y);  bu.p[7] = cvt2bf(f3.z,  f3.w);
            c = __builtin_amdgcn_wmma_f32_16x16x32_bf16(false, au.v, false, bu.v,
                                                        (short)0, c, false, false);
        }

        #pragma unroll
        for (int r = 0; r < 8; ++r) {
            float v = c[r];
            v = fmaxf(v, __shfl_xor(v, 1, 32));
            v = fmaxf(v, __shfl_xor(v, 2, 32));
            v = fmaxf(v, __shfl_xor(v, 4, 32));
            v = fmaxf(v, __shfl_xor(v, 8, 32));
            mw[wave * 16 + r + 8 * hi] = v;
        }
        __syncthreads();

        float mnew = m_run;
        #pragma unroll
        for (int w2 = 0; w2 < WAVES; ++w2) mnew = fmaxf(mnew, mw[w2 * 16 + ln]);
        const float alpha = fast_exp2(m_run - mnew);
        m_run = mnew;
        bcst[wave * 32 + ln]      = mnew;
        bcst[wave * 32 + 16 + ln] = alpha;
        float mnb[8], alb[8];
        {
            const float4 x0 = *(const float4*)(bcst + wave * 32 + 8 * hi);
            const float4 x1 = *(const float4*)(bcst + wave * 32 + 8 * hi + 4);
            mnb[0]=x0.x; mnb[1]=x0.y; mnb[2]=x0.z; mnb[3]=x0.w;
            mnb[4]=x1.x; mnb[5]=x1.y; mnb[6]=x1.z; mnb[7]=x1.w;
            const float4 y0 = *(const float4*)(bcst + wave * 32 + 16 + 8 * hi);
            const float4 y1 = *(const float4*)(bcst + wave * 32 + 16 + 8 * hi + 4);
            alb[0]=y0.x; alb[1]=y0.y; alb[2]=y0.z; alb[3]=y0.w;
            alb[4]=y1.x; alb[5]=y1.y; alb[6]=y1.z; alb[7]=y1.w;
        }

        #pragma unroll
        for (int r = 0; r < 8; ++r) {
            const float p = fast_exp2(c[r] - mnb[r]);
            Pb[(r + 8 * hi) * PS_STRIDE + wave * 16 + ln] = f2bf(p);
            float v = p;
            v += __shfl_xor(v, 1, 32);
            v += __shfl_xor(v, 2, 32);
            v += __shfl_xor(v, 4, 32);
            v += __shfl_xor(v, 8, 32);
            sw[wave * 16 + r + 8 * hi] = v;
        }
        #pragma unroll
        for (int t = 0; t < 8; ++t) {
            #pragma unroll
            for (int r = 0; r < 8; ++r) o[t][r] *= alb[r];
        }
        __syncthreads();

        float lsum = 0.f;
        #pragma unroll
        for (int w2 = 0; w2 < WAVES; ++w2) lsum += sw[w2 * 16 + ln];
        l_run = alpha * l_run + lsum;

        #pragma unroll
        for (int kf = 0; kf < 4; ++kf) {
            union { v8bf h[2]; v16bf v; } au;
            au.h[0] = *(const v8bf*)(Pb + ln * PS_STRIDE + kf * 32 + 16 * hi);
            au.h[1] = *(const v8bf*)(Pb + ln * PS_STRIDE + kf * 32 + 16 * hi + 8);
            const float* vp = V + (size_t)(nb + kf * 32 + 16 * hi) * DMODEL + wave * DSLICE + ln;
            #pragma unroll
            for (int t = 0; t < 8; ++t) {
                float fv[16];
                #pragma unroll
                for (int e = 0; e < 16; ++e)
                    fv[e] = vp[(size_t)e * DMODEL + t * 16];
                union { v2bf p[8]; v16bf v; } bu;
                #pragma unroll
                for (int e = 0; e < 8; ++e)
                    bu.p[e] = cvt2bf(fv[2 * e], fv[2 * e + 1]);
                o[t] = __builtin_amdgcn_wmma_f32_16x16x32_bf16(false, au.v, false, bu.v,
                                                               (short)0, o[t], false, false);
            }
        }
    }

    const float inv = 1.0f / l_run;
    bcst[wave * 32 + ln] = inv;
    float invb[8];
    {
        const float4 z0 = *(const float4*)(bcst + wave * 32 + 8 * hi);
        const float4 z1 = *(const float4*)(bcst + wave * 32 + 8 * hi + 4);
        invb[0]=z0.x; invb[1]=z0.y; invb[2]=z0.z; invb[3]=z0.w;
        invb[4]=z1.x; invb[5]=z1.y; invb[6]=z1.z; invb[7]=z1.w;
    }
    #pragma unroll
    for (int t = 0; t < 8; ++t) {
        #pragma unroll
        for (int r = 0; r < 8; ++r) {
            O[(size_t)(qbase + r + 8 * hi) * DMODEL + wave * DSLICE + t * 16 + ln]
                = o[t][r] * invb[r];
        }
    }
}

extern "C" void kernel_launch(void* const* d_in, const int* in_sizes, int n_in,
                              void* d_out, int out_size, void* d_ws, size_t ws_size,
                              hipStream_t stream) {
    (void)in_sizes; (void)n_in; (void)out_size;
    const float* q = (const float*)d_in[0];
    const float* k = (const float*)d_in[1];
    const float* v = (const float*)d_in[2];
    float* out = (float*)d_out;

    const size_t need = (size_t)2 * N_CTX * DMODEL * sizeof(__bf16);   // 32 MB
    if (d_ws != nullptr && ws_size >= need) {
        __bf16* Kb = (__bf16*)d_ws;
        __bf16* Vt = Kb + (size_t)N_CTX * DMODEL;
        convert_k_bf16<<<dim3((N_CTX * DMODEL) / (256 * 8)), dim3(256), 0, stream>>>(k, Kb);
        transpose_v_bf16<<<dim3((N_CTX / 64) * (DMODEL / 64)), dim3(256), 0, stream>>>(v, Vt);
        fattn_fwd_bf16<<<dim3(N_CTX / BM), dim3(256), 0, stream>>>(q, Kb, Vt, out);
    } else {
        fattn_fwd_f32<<<dim3(N_CTX / BM), dim3(256), 0, stream>>>(q, k, v, out);
    }
}
